// ScaledMultiHeadAttention_43215960933211
// MI455X (gfx1250) — matrix operand order repaired and checked by reading, not compiled, not run on hardware
//
#include <hip/hip_runtime.h>
#include <hip/hip_bf16.h>

#define BB 4
#define SS 1024
#define DD 128
#define HH 8
#define HDIM 16
#define NREL 1001   // 2*500+1

typedef _Float16 v16h __attribute__((ext_vector_type(16)));
typedef _Float16 v8h  __attribute__((ext_vector_type(8)));
typedef float    v8f  __attribute__((ext_vector_type(8)));

// ---- fp32 -> fp16 conversion (inputs, weights, emb table) ----
__global__ __launch_bounds__(256) void cvt_kernel(const float* __restrict__ s,
                                                  _Float16* __restrict__ d, int n) {
  int i = blockIdx.x * 256 + threadIdx.x;
  if (i < n) d[i] = (_Float16)s[i];
}

// A-matrix lane load: 16x32 f16 tile, lane holds row `row`, K-half by `hi`.
// elements 0..7 = row[k0 + 8*hi .. +7], elements 8..15 = row[k0+16+8*hi .. +7]
static __device__ __forceinline__ v16h load_a16(const _Float16* __restrict__ row,
                                                int k0, int hi) {
  v16h a;
  v8h lo = *(const v8h*)(row + k0 + (hi << 3));
  v8h up = *(const v8h*)(row + k0 + 16 + (hi << 3));
#pragma unroll
  for (int j = 0; j < 8; ++j) { a[j] = lo[j]; a[8 + j] = up[j]; }
  return a;
}

// ---- Q/K/V projections on the WMMA pipe ----
// One wave per (which, b, 16-row s-tile, 16-col d-tile); K=128 in 4 chunks.
__global__ __launch_bounds__(256) void proj_wmma(
    const _Float16* __restrict__ Xq, const _Float16* __restrict__ Xk,
    const _Float16* __restrict__ Xv,
    const _Float16* __restrict__ Wqh, const _Float16* __restrict__ Wkh,
    const _Float16* __restrict__ Wvh,
    const float* __restrict__ bq, const float* __restrict__ bk,
    const float* __restrict__ bv,
    _Float16* __restrict__ Qh, _Float16* __restrict__ Kh, _Float16* __restrict__ Vt)
{
  const int lane = threadIdx.x & 31;
  const int wid  = threadIdx.x >> 5;
  const int tile = blockIdx.x * 8 + wid;      // 0 .. 6143
  const int which = tile >> 11;               // 0=Q 1=K 2=V
  const int rem = tile & 2047;
  const int d0 = (rem & 7) << 4;
  const int st = rem >> 3;
  const int s0 = (st & 63) << 4;
  const int b  = st >> 6;
  const int hi = lane >> 4;
  const int n  = lane & 15;

  const _Float16* X = which == 0 ? Xq : (which == 1 ? Xk : Xv);
  const _Float16* W = which == 0 ? Wqh : (which == 1 ? Wkh : Wvh);
  const float* bias = which == 0 ? bq : (which == 1 ? bk : bv);

  const _Float16* xrow = X + ((size_t)b * SS + s0 + n) * DD;  // A row per lane
  const _Float16* wrow = W + (size_t)(d0 + n) * DD;           // B column source

  v8f acc;
#pragma unroll
  for (int r = 0; r < 8; ++r) acc[r] = 0.f;
#pragma unroll
  for (int k0 = 0; k0 < DD; k0 += 32) {
    v16h a  = load_a16(xrow, k0, hi);
    v16h bm = load_a16(wrow, k0, hi);  // same k order as A
    acc = __builtin_amdgcn_wmma_f32_16x16x32_f16(false, a, false, bm, (short)0, acc, false, false);
  }

  const float bb = bias[d0 + n];
  const int h = d0 >> 4;
#pragma unroll
  for (int r = 0; r < 8; ++r) {
    const int s = s0 + r + (hi << 3);
    const float vl = acc[r] + bb;
    if (which == 0)
      Qh[(((size_t)b * HH + h) * SS + s) * HDIM + n] = (_Float16)vl;
    else if (which == 1)
      Kh[(((size_t)b * HH + h) * SS + s) * HDIM + n] = (_Float16)(vl * 0.25f);
    else
      Vt[(((size_t)b * HH + h) * HDIM + n) * SS + s] = (_Float16)vl;
  }
}

// ---- Flash-attention core (unchanged math; ctx now stored f16) ----
__global__ __launch_bounds__(256) void attn_kernel(
    const _Float16* __restrict__ Qh, const _Float16* __restrict__ Kh,
    const _Float16* __restrict__ Vt, const _Float16* __restrict__ Eh,
    _Float16* __restrict__ ctxh)
{
  __shared__ alignas(32) _Float16 smem[8][256];

  const int lane = threadIdx.x & 31;
  const int wid  = threadIdx.x >> 5;
  const int tile = blockIdx.x * 8 + wid;      // 0 .. 2047
  const int q0 = (tile & 63) << 4;
  const int h  = (tile >> 6) & 7;
  const int b  = tile >> 9;

  const int hi   = lane >> 4;
  const int n    = lane & 15;
  const int hi16 = hi << 4;

  v16h aq;
#pragma unroll
  for (int i = 0; i < 16; ++i) aq[i] = (_Float16)0;
  {
    if (hi == 0) aq = *(const v16h*)(Qh + ((((size_t)b * HH + h) * SS + q0 + n) << 4));
  }

  v8f co;
  float run_max[8], run_sum[8];
#pragma unroll
  for (int r = 0; r < 8; ++r) { co[r] = 0.f; run_max[r] = -1e30f; run_sum[r] = 0.f; }

  const _Float16* Kbase = Kh + ((((size_t)b * HH + h) * SS) << 4);
  const _Float16* Vbase = Vt + (((size_t)b * HH + h) * HDIM + n) * SS;
  const _Float16* Ebase = Eh + h * HDIM;

  for (int k0 = 0; k0 < SS; k0 += 16) {
    v16h bkm, tb1, tb2;
#pragma unroll
    for (int i = 0; i < 16; ++i) { bkm[i] = (_Float16)0; tb1[i] = (_Float16)0; tb2[i] = (_Float16)0; }
    if (hi == 0) {
      bkm = *(const v16h*)(Kbase + ((size_t)(k0 + n) << 4));
      int dd = k0 - q0 + n;
      int r1 = dd - 15; r1 = r1 < -500 ? -500 : (r1 > 500 ? 500 : r1);
      int r2 = dd;      r2 = r2 < -500 ? -500 : (r2 > 500 ? 500 : r2);
      tb1 = *(const v16h*)(Ebase + (size_t)(r1 + 500) * DD);
      tb2 = *(const v16h*)(Ebase + (size_t)(r2 + 500) * DD);
      if (k0 + 16 < SS)
        __builtin_prefetch(Kbase + ((size_t)(k0 + 16 + n) << 4), 0, 1);
    }

    v8f z;
#pragma unroll
    for (int r = 0; r < 8; ++r) z[r] = 0.f;
    v8f sc = __builtin_amdgcn_wmma_f32_16x16x32_f16(false, aq, false, bkm, (short)0, z, false, false);
    v8f c1 = __builtin_amdgcn_wmma_f32_16x16x32_f16(false, aq, false, tb1, (short)0, z, false, false);
    v8f c2 = __builtin_amdgcn_wmma_f32_16x16x32_f16(false, aq, false, tb2, (short)0, z, false, false);

#pragma unroll
    for (int r = 0; r < 8; ++r) {
      const int m = r + (hi << 3);
      const int delta = n - m;
      const int l1 = ((delta < 0) ? (delta + 15) : 0) + hi16;
      const int l2 = ((delta >= 0) ? delta : 0) + hi16;
      float v1 = __shfl(c1[r], l1, 32);
      float v2 = __shfl(c2[r], l2, 32);
      float s = sc[r] + ((delta >= 0) ? v2 : v1);

      float mx = s;
#pragma unroll
      for (int off = 8; off >= 1; off >>= 1) mx = fmaxf(mx, __shfl_xor(mx, off, 16));
      float nm = fmaxf(run_max[r], mx);
      float p  = __expf(s - nm);
      float ps = p;
#pragma unroll
      for (int off = 8; off >= 1; off >>= 1) ps += __shfl_xor(ps, off, 16);
      float alpha = __expf(run_max[r] - nm);
      run_sum[r] = run_sum[r] * alpha + ps;
      run_max[r] = nm;
      co[r] *= alpha;
      smem[wid][m * 16 + n] = (_Float16)p;
    }
    __builtin_amdgcn_wave_barrier();

    v16h pa, bvm;
#pragma unroll
    for (int i = 0; i < 16; ++i) { pa[i] = (_Float16)0; bvm[i] = (_Float16)0; }
    {
      if (hi == 0) pa = *(const v16h*)&smem[wid][n * 16];
    }
    if (hi == 0) bvm = *(const v16h*)(Vbase + k0);
    co = __builtin_amdgcn_wmma_f32_16x16x32_f16(false, pa, false, bvm, (short)0, co, false, false);
    __builtin_amdgcn_wave_barrier();
  }

#pragma unroll
  for (int r = 0; r < 8; ++r) {
    const int m = r + (hi << 3);
    ctxh[((size_t)b * SS + q0 + m) * DD + h * HDIM + n] = (_Float16)(co[r] / run_sum[r]);
  }
}

// ---- Output projection on the WMMA pipe: out = ctx @ Wo^T + bo (fp32 out) ----
__global__ __launch_bounds__(256) void out_wmma(
    const _Float16* __restrict__ Ch, const _Float16* __restrict__ Woh,
    const float* __restrict__ bo, float* __restrict__ out)
{
  const int lane = threadIdx.x & 31;
  const int wid  = threadIdx.x >> 5;
  const int tile = blockIdx.x * 8 + wid;      // 0 .. 2047
  const int d0 = (tile & 7) << 4;
  const int st = tile >> 3;
  const int s0 = (st & 63) << 4;
  const int b  = st >> 6;
  const int hi = lane >> 4;
  const int n  = lane & 15;

  const _Float16* xrow = Ch + ((size_t)b * SS + s0 + n) * DD;
  const _Float16* wrow = Woh + (size_t)(d0 + n) * DD;

  v8f acc;
#pragma unroll
  for (int r = 0; r < 8; ++r) acc[r] = 0.f;
#pragma unroll
  for (int k0 = 0; k0 < DD; k0 += 32) {
    v16h a  = load_a16(xrow, k0, hi);
    v16h bm = load_a16(wrow, k0, hi);
    acc = __builtin_amdgcn_wmma_f32_16x16x32_f16(false, a, false, bm, (short)0, acc, false, false);
  }

  const float bb = bo[d0 + n];
#pragma unroll
  for (int r = 0; r < 8; ++r) {
    const int s = s0 + r + (hi << 3);
    out[((size_t)b * SS + s) * DD + d0 + n] = acc[r] + bb;
  }
}

extern "C" void kernel_launch(void* const* d_in, const int* in_sizes, int n_in,
                              void* d_out, int out_size, void* d_ws, size_t ws_size,
                              hipStream_t stream) {
  (void)in_sizes; (void)n_in; (void)out_size; (void)ws_size;
  const float* query = (const float*)d_in[0];
  const float* key   = (const float*)d_in[1];
  const float* value = (const float*)d_in[2];
  const float* Wq    = (const float*)d_in[3];
  const float* bq    = (const float*)d_in[4];
  const float* Wk    = (const float*)d_in[5];
  const float* bk    = (const float*)d_in[6];
  const float* Wv    = (const float*)d_in[7];
  const float* bv    = (const float*)d_in[8];
  const float* Wo    = (const float*)d_in[9];
  const float* bo    = (const float*)d_in[10];
  const float* emb   = (const float*)d_in[11];

  char* ws = (char*)d_ws;
  const size_t MB = 1u << 20;
  _Float16* Qh   = (_Float16*)(ws);                        // 1 MB
  _Float16* Kh   = (_Float16*)(ws + 1 * MB);               // 1 MB
  _Float16* Vt   = (_Float16*)(ws + 2 * MB);               // 1 MB
  _Float16* Eh   = (_Float16*)(ws + 3 * MB);               // 256 KB (pad 512K)
  _Float16* Ch   = (_Float16*)(ws + 3 * MB + 512 * 1024);  // 1 MB (ctx f16)
  _Float16* Xqh  = (_Float16*)(ws + 4 * MB + 512 * 1024);  // 1 MB
  _Float16* Xkh  = (_Float16*)(ws + 5 * MB + 512 * 1024);  // 1 MB
  _Float16* Xvh  = (_Float16*)(ws + 6 * MB + 512 * 1024);  // 1 MB
  _Float16* Wqh  = (_Float16*)(ws + 7 * MB + 512 * 1024);          // 32 KB
  _Float16* Wkh  = (_Float16*)(ws + 7 * MB + 512 * 1024 + 32768);  // 32 KB
  _Float16* Wvh  = (_Float16*)(ws + 7 * MB + 512 * 1024 + 65536);  // 32 KB
  _Float16* Woh  = (_Float16*)(ws + 7 * MB + 512 * 1024 + 98304);  // 32 KB

  const int NX = BB * SS * DD;   // 524288
  const int NW = DD * DD;        // 16384
  cvt_kernel<<<NX / 256, 256, 0, stream>>>(query, Xqh, NX);
  cvt_kernel<<<NX / 256, 256, 0, stream>>>(key,   Xkh, NX);
  cvt_kernel<<<NX / 256, 256, 0, stream>>>(value, Xvh, NX);
  cvt_kernel<<<NW / 256, 256, 0, stream>>>(Wq, Wqh, NW);
  cvt_kernel<<<NW / 256, 256, 0, stream>>>(Wk, Wkh, NW);
  cvt_kernel<<<NW / 256, 256, 0, stream>>>(Wv, Wvh, NW);
  cvt_kernel<<<NW / 256, 256, 0, stream>>>(Wo, Woh, NW);
  cvt_kernel<<<(NREL * DD + 255) / 256, 256, 0, stream>>>(emb, Eh, NREL * DD);

  proj_wmma<<<768, 256, 0, stream>>>(Xqh, Xkh, Xvh, Wqh, Wkh, Wvh, bq, bk, bv, Qh, Kh, Vt);
  attn_kernel<<<256, 256, 0, stream>>>(Qh, Kh, Vt, Eh, Ch);
  out_wmma<<<256, 256, 0, stream>>>(Ch, Woh, bo, (float*)d_out);
}
